// UserEncoder_63702954934945
// MI455X (gfx1250) — matrix operand order repaired and checked by reading, not compiled, not run on hardware
//
#include <hip/hip_runtime.h>
#include <hip/hip_bf16.h>

// ---------------------------------------------------------------------------
// Fastformer encoder for MI455X (gfx1250, wave32, WMMA).
// GEMMs: bf16 WMMA (v_wmma_f32_16x16x32_bf16), f32 accumulate,
// double-buffered LDS staged with global_load_async_to_lds_b128 (ASYNCcnt).
// ---------------------------------------------------------------------------

typedef __bf16 bf16;
typedef __attribute__((ext_vector_type(16))) __bf16 v16bf;
typedef __attribute__((ext_vector_type(8)))  __bf16 v8bf;
typedef __attribute__((ext_vector_type(8)))  float  v8f;

#define Lc     4
#define Bc     64
#define Nc     128
#define Dc     1024
#define Hc     16
#define HDc    64
#define INTERc 4096
#define Mc     (Bc * Nc)   // 8192 rows

// GEMM tiling
#define BM   128
#define BN   128
#define BK   64
#define KPAD 72            // LDS row stride in elements (144B, 16B-aligned)

// CDNA5 async global->LDS staging (set to 0 to fall back to sync staging)
#define USE_ASYNC 1

__device__ inline bf16 f2bf(float f) { return (bf16)f; }

// ---------------------------------------------------------------------------
// fp32 -> bf16 elementwise conversion (activations)
// ---------------------------------------------------------------------------
__global__ void cvt_bf16_kernel(const float* __restrict__ in,
                                bf16* __restrict__ out, int n) {
  int i = blockIdx.x * blockDim.x + threadIdx.x;
  if (i < n) out[i] = f2bf(in[i]);
}

// ---------------------------------------------------------------------------
// Weight prep: fp32 W[K][N] -> bf16 W^T[N][K], LDS-tiled 32x32 transpose.
// Grid: (N/32, K/32), 256 threads.
// ---------------------------------------------------------------------------
__global__ __launch_bounds__(256) void wt_transpose_cvt_kernel(
    const float* __restrict__ in, bf16* __restrict__ out, int K, int N) {
  __shared__ float tile[32][33];
  int bx = blockIdx.x * 32;  // N direction
  int by = blockIdx.y * 32;  // K direction
  int tx = threadIdx.x & 31;
  int ty = threadIdx.x >> 5;  // 0..7
#pragma unroll
  for (int r = ty; r < 32; r += 8)
    tile[r][tx] = in[(size_t)(by + r) * N + bx + tx];
  __syncthreads();
#pragma unroll
  for (int r = ty; r < 32; r += 8)
    out[(size_t)(bx + r) * K + by + tx] = f2bf(tile[tx][r]);
}

// ---------------------------------------------------------------------------
// Tile staging: copy BM x BK bf16 rows (contiguous in K) into LDS.
// Each thread moves 4 chunks of 8 bf16 (16B) per matrix.
// ---------------------------------------------------------------------------
__device__ inline void stage_tile(const bf16* __restrict__ gA,
                                  const bf16* __restrict__ gB,
                                  bf16 (*__restrict__ As)[KPAD],
                                  bf16 (*__restrict__ Bs)[KPAD],
                                  int tid, int K, int k0) {
#pragma unroll
  for (int it = 0; it < 4; ++it) {
    int chunk = tid + it * 256;         // 0..1023
    int r = chunk >> 3;                 // 0..127
    int c = (chunk & 7) << 3;           // 0..56
    const bf16* pa = gA + (size_t)r * K + k0 + c;
    const bf16* pb = gB + (size_t)r * K + k0 + c;
#if USE_ASYNC
    // LDS byte offset = low 32 bits of flat shared address (aperture stripped).
    uint32_t la = (uint32_t)(uintptr_t)&As[r][c];
    uint32_t lb = (uint32_t)(uintptr_t)&Bs[r][c];
    asm volatile("global_load_async_to_lds_b128 %0, %1, off"
                 :: "v"(la), "v"(pa) : "memory");
    asm volatile("global_load_async_to_lds_b128 %0, %1, off"
                 :: "v"(lb), "v"(pb) : "memory");
#else
    *reinterpret_cast<uint4*>(&As[r][c]) = *reinterpret_cast<const uint4*>(pa);
    *reinterpret_cast<uint4*>(&Bs[r][c]) = *reinterpret_cast<const uint4*>(pb);
#endif
  }
}

// ---------------------------------------------------------------------------
// C[M,N] = A[M,K](bf16, row-major) @ W (given as Bt = W^T[N][K] bf16)
// Epilogue (compile-time): v = acc + bias[n]; [v += res]; [v = gelu(v)];
//                          [Cf = v]; [Cbf = bf16(v)]
// 256 threads = 8 waves (4M x 2N); wave tile 32x64 = 2x4 WMMA accumulators.
// ---------------------------------------------------------------------------
template <bool HAS_RES, bool DO_GELU, bool WF32, bool WBF16>
__global__ __launch_bounds__(256) void wmma_gemm_bf16(
    const bf16* __restrict__ A, const bf16* __restrict__ Bt,
    const float* __restrict__ bias, const float* __restrict__ res,
    float* __restrict__ Cf, bf16* __restrict__ Cbf,
    int M, int N, int K)
{
  __shared__ bf16 As[2][BM][KPAD];
  __shared__ bf16 Bs[2][BM][KPAD];

  const int tid  = threadIdx.x;
  const int lane = tid & 31;
  const int wid  = tid >> 5;     // 0..7
  const int wr   = wid & 3;      // wave row (M)
  const int wc   = wid >> 2;     // wave col (N)
  const int bm   = blockIdx.y * BM;
  const int bn   = blockIdx.x * BN;

  const bf16* gA = A  + (size_t)bm * K;
  const bf16* gB = Bt + (size_t)bn * K;

  v8f acc[2][4] = {};

  stage_tile(gA, gB, As[0], Bs[0], tid, K, 0);

  const int ntiles = K / BK;
  for (int t = 0; t < ntiles; ++t) {
    const int cur = t & 1;
    if (t + 1 < ntiles) {
      stage_tile(gA, gB, As[cur ^ 1], Bs[cur ^ 1], tid, K, (t + 1) * BK);
#if USE_ASYNC
      asm volatile("s_wait_asynccnt 8" ::: "memory");  // tile t's 8 done
#endif
    } else {
#if USE_ASYNC
      asm volatile("s_wait_asynccnt 0" ::: "memory");
#endif
    }
    __syncthreads();

#pragma unroll
    for (int ks = 0; ks < BK; ks += 32) {
      // A frag: lane L: row = base+(L&15); elems 0..15 = K kb..kb+15,
      // kb = (L>=16)?16:0
      v16bf afrag[2];
      {
        int arow = wr * 32 + (lane & 15);
        int akb  = ks + ((lane >> 4) << 4);
#pragma unroll
        for (int mt = 0; mt < 2; ++mt) {
          const v8bf* p =
              reinterpret_cast<const v8bf*>(&As[cur][arow + mt * 16][akb]);
          union { v16bf v; v8bf h[2]; } u;
          u.h[0] = p[0];
          u.h[1] = p[1];
          afrag[mt] = u.v;
        }
      }
      // B frag: lane L: col n = (L&15); K = (L>=16?16:0)..+15 contiguous
      v16bf bfrag[4];
      {
        int brow = wc * 64 + (lane & 15);
        int bkb  = ks + ((lane >> 4) << 4);
#pragma unroll
        for (int nt = 0; nt < 4; ++nt) {
          const v8bf* p =
              reinterpret_cast<const v8bf*>(&Bs[cur][brow + nt * 16][bkb]);
          union { v16bf v; v8bf h[2]; } u;
          u.h[0] = p[0];
          u.h[1] = p[1];
          bfrag[nt] = u.v;
        }
      }
#pragma unroll
      for (int mt = 0; mt < 2; ++mt)
#pragma unroll
        for (int nt = 0; nt < 4; ++nt)
          acc[mt][nt] = __builtin_amdgcn_wmma_f32_16x16x32_bf16(
              false, afrag[mt], false, bfrag[nt],
              (short)0, acc[mt][nt], false, false);
    }
    __syncthreads();
  }

  // Epilogue. C/D layout: n = (lane&15); VGPR g -> m = g + (lane>=16?8:0)
#pragma unroll
  for (int mt = 0; mt < 2; ++mt) {
#pragma unroll
    for (int nt = 0; nt < 4; ++nt) {
      int n = bn + wc * 64 + nt * 16 + (lane & 15);
      float bv = bias[n];
#pragma unroll
      for (int g = 0; g < 8; ++g) {
        int m = bm + wr * 32 + mt * 16 + ((lane >> 4) << 3) + g;
        float v = acc[mt][nt][g] + bv;
        if constexpr (HAS_RES) v += res[(size_t)m * N + n];
        if constexpr (DO_GELU)
          v = 0.5f * v * (1.0f + erff(v * 0.70710678118654752f));
        if constexpr (WF32)  Cf [(size_t)m * N + n] = v;
        if constexpr (WBF16) Cbf[(size_t)m * N + n] = f2bf(v);
      }
    }
  }
}

// ---------------------------------------------------------------------------
// Additive attention: per (b,h) block (128 thr = T):
//   s[t] = dot(v[b,t,h*HD:], hw[h]) / sqrt(HD); alpha = softmax_t(s)
//   g[b, h*HD+d] = sum_t alpha[t] * v[b,t,h*HD+d]
// ---------------------------------------------------------------------------
__global__ __launch_bounds__(128) void attn_reduce_kernel(
    const float* __restrict__ v, const float* __restrict__ hw,
    float* __restrict__ g)
{
  int bh = blockIdx.x;
  int b = bh / Hc, h = bh % Hc;
  int t = threadIdx.x;
  const float* row = v + ((size_t)b * Nc + t) * Dc + h * HDc;
  const float* w = hw + h * HDc;
  float s = 0.0f;
#pragma unroll 8
  for (int d = 0; d < HDc; ++d) s += row[d] * w[d];
  s *= 0.125f;  // 1/sqrt(64)

  __shared__ float sc[Nc];
  __shared__ float red[Nc];
  red[t] = s;
  __syncthreads();
  for (int off = 64; off > 0; off >>= 1) {
    if (t < off) red[t] = fmaxf(red[t], red[t + off]);
    __syncthreads();
  }
  float mx = red[0];
  __syncthreads();
  float e = __expf(s - mx);
  sc[t] = e;
  red[t] = e;
  __syncthreads();
  for (int off = 64; off > 0; off >>= 1) {
    if (t < off) red[t] += red[t + off];
    __syncthreads();
  }
  float inv_denom = 1.0f / red[0];
  __syncthreads();

  if (t < HDc) {
    const float* basep = v + (size_t)b * Nc * Dc + h * HDc + t;
    float a = 0.0f;
    for (int tt = 0; tt < Nc; ++tt) a += sc[tt] * basep[(size_t)tt * Dc];
    g[(size_t)b * Dc + h * HDc + t] = a * inv_denom;
  }
}

// ---------------------------------------------------------------------------
// out = a * broadcast(g over tokens); optional fp32 / bf16 outputs
// ---------------------------------------------------------------------------
template <bool WF32, bool WBF16>
__global__ void mul_bcast_kernel(const float* __restrict__ a,
                                 const float* __restrict__ g,
                                 float* __restrict__ outf,
                                 bf16* __restrict__ outbf, int total)
{
  int i = blockIdx.x * blockDim.x + threadIdx.x;
  if (i >= total) return;
  int b = i / (Nc * Dc);
  int d = i % Dc;
  float v = a[i] * g[(size_t)b * Dc + d];
  if constexpr (WF32)  outf[i]  = v;
  if constexpr (WBF16) outbf[i] = f2bf(v);
}

// ---------------------------------------------------------------------------
// LayerNorm over last dim (Dc); one block (256 thr) per row
// ---------------------------------------------------------------------------
__global__ __launch_bounds__(256) void layernorm_kernel(
    const float* __restrict__ in, const float* __restrict__ gm,
    const float* __restrict__ bt, float* __restrict__ outf,
    bf16* __restrict__ outbf)
{
  int row = blockIdx.x, t = threadIdx.x;
  const float* x = in + (size_t)row * Dc;
  float s = 0.0f, s2 = 0.0f;
  for (int d = t; d < Dc; d += 256) { float v = x[d]; s += v; s2 += v * v; }
  __shared__ float r1[256], r2[256];
  r1[t] = s; r2[t] = s2;
  __syncthreads();
  for (int off = 128; off > 0; off >>= 1) {
    if (t < off) { r1[t] += r1[t + off]; r2[t] += r2[t + off]; }
    __syncthreads();
  }
  float mean = r1[0] * (1.0f / Dc);
  float var  = r2[0] * (1.0f / Dc) - mean * mean;
  float inv  = rsqrtf(var + 1e-12f);
  for (int d = t; d < Dc; d += 256) {
    float v = (x[d] - mean) * inv * gm[d] + bt[d];
    outf [(size_t)row * Dc + d] = v;
    outbf[(size_t)row * Dc + d] = f2bf(v);
  }
}

// ---------------------------------------------------------------------------
// scores[row] = sum_d tanh(s1[row,d]) * pq[d]; one block per row
// ---------------------------------------------------------------------------
__global__ __launch_bounds__(256) void tanh_dot_kernel(
    const float* __restrict__ s1, const float* __restrict__ pq,
    float* __restrict__ scores)
{
  int row = blockIdx.x, t = threadIdx.x;
  const float* x = s1 + (size_t)row * Dc;
  float s = 0.0f;
  for (int d = t; d < Dc; d += 256) s += tanhf(x[d]) * pq[d];
  __shared__ float r[256];
  r[t] = s;
  __syncthreads();
  for (int off = 128; off > 0; off >>= 1) {
    if (t < off) r[t] += r[t + off];
    __syncthreads();
  }
  if (t == 0) scores[row] = r[0];
}

// ---------------------------------------------------------------------------
// Final pooling: per batch b: w = softmax_n(scores[b,:]);
// out[b,d] = sum_n w[n] * hfin[b,n,d]
// ---------------------------------------------------------------------------
__global__ __launch_bounds__(256) void pool_finish_kernel(
    const float* __restrict__ scores, const float* __restrict__ hfin,
    float* __restrict__ out)
{
  int b = blockIdx.x, t = threadIdx.x;
  __shared__ float w[Nc];
  __shared__ float r[256];
  float s = (t < Nc) ? scores[b * Nc + t] : -1e30f;
  r[t] = s;
  __syncthreads();
  for (int off = 128; off > 0; off >>= 1) {
    if (t < off) r[t] = fmaxf(r[t], r[t + off]);
    __syncthreads();
  }
  float mx = r[0];
  __syncthreads();
  float e = (t < Nc) ? __expf(s - mx) : 0.0f;
  r[t] = e;
  __syncthreads();
  for (int off = 128; off > 0; off >>= 1) {
    if (t < off) r[t] += r[t + off];
    __syncthreads();
  }
  float inv_denom = 1.0f / r[0];
  __syncthreads();
  if (t < Nc) w[t] = e * inv_denom;
  __syncthreads();
  for (int d = t; d < Dc; d += 256) {
    float a = 0.0f;
    for (int n = 0; n < Nc; ++n)
      a += w[n] * hfin[((size_t)b * Nc + n) * Dc + d];
    out[(size_t)b * Dc + d] = a;
  }
}

// ---------------------------------------------------------------------------
// Host orchestration
// ---------------------------------------------------------------------------
extern "C" void kernel_launch(void* const* d_in, const int* in_sizes, int n_in,
                              void* d_out, int out_size, void* d_ws, size_t ws_size,
                              hipStream_t stream) {
  const float* x     = (const float*)d_in[0];
  const float* wq_w  = (const float*)d_in[1];
  const float* wq_b  = (const float*)d_in[2];
  const float* wk_w  = (const float*)d_in[3];
  const float* wk_b  = (const float*)d_in[4];
  const float* hwq   = (const float*)d_in[5];
  const float* hwk   = (const float*)d_in[6];
  const float* tr_w  = (const float*)d_in[7];
  const float* tr_b  = (const float*)d_in[8];
  const float* so_w  = (const float*)d_in[9];
  const float* so_b  = (const float*)d_in[10];
  const float* ln1_g = (const float*)d_in[11];
  const float* ln1_b = (const float*)d_in[12];
  const float* int_w = (const float*)d_in[13];
  const float* int_b = (const float*)d_in[14];
  const float* out_w = (const float*)d_in[15];
  const float* out_b = (const float*)d_in[16];
  const float* ln2_g = (const float*)d_in[17];
  const float* ln2_b = (const float*)d_in[18];
  const float* pool_w = (const float*)d_in[19];
  const float* pool_b = (const float*)d_in[20];
  const float* pool_q = (const float*)d_in[21];
  float* outp = (float*)d_out;

  char* base = (char*)d_ws;
  size_t off = 0;
  auto alloc = [&](size_t bytes) -> char* {
    bytes = (bytes + 255) & ~(size_t)255;
    char* p = base + off;
    off += bytes;
    return p;
  };

  const size_t MD = (size_t)Mc * Dc;
  const size_t MI = (size_t)Mc * INTERc;

  bf16*  act_bf = (bf16*)alloc(MD * 2);
  bf16*  a0_bf  = (bf16*)alloc(MD * 2);
  bf16*  a_bf   = (bf16*)alloc(MD * 2);
  bf16*  u_bf   = (bf16*)alloc(MD * 2);
  bf16*  h_bf   = (bf16*)alloc(MI * 2);
  bf16*  w_bf   = (bf16*)alloc((size_t)Dc * INTERc * 2);  // W^T scratch
  float* qf     = (float*)alloc(MD * 4);
  float* kf     = (float*)alloc(MD * 4);
  float* buf1   = (float*)alloc(MD * 4);
  float* buf2   = (float*)alloc(MD * 4);
  float* af     = (float*)alloc(MD * 4);
  float* actf   = (float*)alloc(MD * 4);
  float* gq     = (float*)alloc((size_t)Bc * Dc * 4);
  float* gk     = (float*)alloc((size_t)Bc * Dc * 4);
  float* scores = (float*)alloc((size_t)Bc * Nc * 4);

  auto cvt = [&](const float* src, bf16* dst, size_t n) {
    cvt_bf16_kernel<<<(unsigned)((n + 255) / 256), 256, 0, stream>>>(src, dst, (int)n);
  };
  // W[K][N] fp32 -> W^T[N][K] bf16
  auto wprep = [&](const float* w, int K, int N) {
    dim3 grid(N / 32, K / 32);
    wt_transpose_cvt_kernel<<<grid, 256, 0, stream>>>(w, w_bf, K, N);
  };
  auto gemm = [&](const bf16* A, const float* bias, const float* res,
                  float* Cf, bf16* Cbf, int M, int N, int K, int gel) {
    dim3 grid(N / BN, M / BM);
    if (res && Cf && Cbf)
      wmma_gemm_bf16<true, false, true, true>
          <<<grid, 256, 0, stream>>>(A, w_bf, bias, res, Cf, Cbf, M, N, K);
    else if (res && Cf)
      wmma_gemm_bf16<true, false, true, false>
          <<<grid, 256, 0, stream>>>(A, w_bf, bias, res, Cf, Cbf, M, N, K);
    else if (gel)
      wmma_gemm_bf16<false, true, false, true>
          <<<grid, 256, 0, stream>>>(A, w_bf, bias, res, Cf, Cbf, M, N, K);
    else
      wmma_gemm_bf16<false, false, true, false>
          <<<grid, 256, 0, stream>>>(A, w_bf, bias, res, Cf, Cbf, M, N, K);
  };

  // initial activation -> bf16
  cvt(x, act_bf, MD);

  for (int i = 0; i < Lc; ++i) {
    // q = x @ wq_w + wq_b
    wprep(wq_w + (size_t)i * Dc * Dc, Dc, Dc);
    gemm(act_bf, wq_b + i * Dc, nullptr, qf, nullptr, Mc, Dc, Dc, 0);
    // gq = additive attention over q
    attn_reduce_kernel<<<Bc * Hc, Nc, 0, stream>>>(qf, hwq + i * Hc * HDc, gq);
    // k = x @ wk_w + wk_b
    wprep(wk_w + (size_t)i * Dc * Dc, Dc, Dc);
    gemm(act_bf, wk_b + i * Dc, nullptr, kf, nullptr, Mc, Dc, Dc, 0);
    // p = k * gq (in place, fp32)
    mul_bcast_kernel<true, false><<<(unsigned)((MD + 255) / 256), 256, 0, stream>>>(
        kf, gq, kf, nullptr, (int)MD);
    // gk = additive attention over p
    attn_reduce_kernel<<<Bc * Hc, Nc, 0, stream>>>(kf, hwk + i * Hc * HDc, gk);
    // u = q * gk (bf16 for GEMM)
    mul_bcast_kernel<false, true><<<(unsigned)((MD + 255) / 256), 256, 0, stream>>>(
        qf, gk, nullptr, u_bf, (int)MD);
    // a0 = u @ tr_w + tr_b + q  (fp32 + bf16)
    wprep(tr_w + (size_t)i * Dc * Dc, Dc, Dc);
    gemm(u_bf, tr_b + i * Dc, qf, buf1, a0_bf, Mc, Dc, Dc, 0);
    // a1 = a0 @ so_w + so_b
    wprep(so_w + (size_t)i * Dc * Dc, Dc, Dc);
    gemm(a0_bf, so_b + i * Dc, nullptr, buf2, nullptr, Mc, Dc, Dc, 0);
    // a = LN1(a1)
    layernorm_kernel<<<Mc, 256, 0, stream>>>(buf2, ln1_g + i * Dc, ln1_b + i * Dc,
                                             af, a_bf);
    // h = gelu(a @ int_w + int_b)  (bf16 only, fused GELU)
    wprep(int_w + (size_t)i * Dc * INTERc, Dc, INTERc);
    gemm(a_bf, int_b + i * INTERc, nullptr, nullptr, h_bf, Mc, INTERc, Dc, 1);
    // o + a = h @ out_w + out_b + a
    wprep(out_w + (size_t)i * INTERc * Dc, INTERc, Dc);
    gemm(h_bf, out_b + i * Dc, af, buf1, nullptr, Mc, Dc, INTERc, 0);
    // x_next = LN2(o + a) -> fp32 + bf16
    layernorm_kernel<<<Mc, 256, 0, stream>>>(buf1, ln2_g + i * Dc, ln2_b + i * Dc,
                                             actf, act_bf);
  }

  // pooling: s1 = h @ pool_w + pool_b
  wprep(pool_w, Dc, Dc);
  gemm(act_bf, pool_b, nullptr, buf2, nullptr, Mc, Dc, Dc, 0);
  tanh_dot_kernel<<<Mc, 256, 0, stream>>>(buf2, pool_q, scores);
  pool_finish_kernel<<<Bc, 256, 0, stream>>>(scores, actf, outp);
}
